// VisionTransformer_67448166416490
// MI455X (gfx1250) — compile-verified, matrix operand order checked by reading, not run on hardware
//
#include <hip/hip_runtime.h>
#include <cstdint>
#include <cstddef>

// ---------------------------------------------------------------------------
// NACLIP ViT forward for MI455X (gfx1250, wave32, WMMA).
// All GEMMs via v_wmma_f32_16x16x32_f16 (f16 in, f32 accumulate).
// Full tiles stage A (and f16 W) with GLOBAL_LOAD_ASYNC_TO_LDS_B128 (ASYNCcnt);
// fp32 weights are converted to f16 while staging tiles into LDS.
// Epilogue configuration (bias/residual/output dtypes) is compile-time.
// ---------------------------------------------------------------------------

typedef _Float16 half_t;
typedef __attribute__((ext_vector_type(16))) _Float16 v16h;
typedef __attribute__((ext_vector_type(8)))  _Float16 v8h;
typedef __attribute__((ext_vector_type(8)))  float    v8f;
typedef __attribute__((ext_vector_type(4)))  float    v4f;

namespace {
constexpr int B_   = 32;
constexpr int NV   = 197;    // valid tokens
constexpr int NTP  = 224;    // padded tokens (multiple of 32)
constexpr int BT   = B_ * NTP;
constexpr int D_   = 768;
constexpr int H_   = 12;
constexpr int HD   = 64;
constexpr int F_   = 3072;
constexpr int OUT_ = 512;
constexpr int L_   = 12;

constexpr int BM = 128, BN = 128, BK = 32;
constexpr int LDT = 40;      // LDS row stride in halves: 80B, 16B-aligned, bank-skewed
}

// Async global->LDS 16B copy (CDNA5 async path, tracked with ASYNCcnt).
// VDST VGPR carries the LDS byte address; generic LDS pointers have the
// LDS offset in the low 32 bits (ISA 10.2 aperture mapping).
__device__ __forceinline__ void async_copy_b128(void* lds_dst, const void* gsrc) {
  uint32_t lofs = (uint32_t)(uintptr_t)lds_dst;
  asm volatile("global_load_async_to_lds_b128 %0, %1, off"
               :: "v"(lofs), "v"(gsrc)
               : "memory");
}
__device__ __forceinline__ void wait_asynccnt0() {
  asm volatile("s_wait_asynccnt 0" ::: "memory");
}

// Load one 16x32 f16 WMMA operand fragment (per ISA 7.12.2 layout) from LDS.
// element e<8  -> k = khi + e        (khi = 0 for lanes 0-15, 8 for lanes 16-31)
// element e>=8 -> k = 16 + khi + e-8
__device__ __forceinline__ v16h ld_frag(const half_t* base) {
  v8h lo = *(const v8h*)(base);
  v8h hi = *(const v8h*)(base + 16);
  v16h r;
#pragma unroll
  for (int i = 0; i < 8; ++i) { r[i] = lo[i]; r[i + 8] = hi[i]; }
  return r;
}

// Generic batched GEMM: C[z] = act(scale * A[z] x W[z]^T + bias) (+ res)
//   A: [M x K] f16 row-major.  W: [N x K] (i.e. row n = output column n),
//   fp32 (converted to f16 in LDS staging) or f16 per template.
//   GUARD=false requires M%BM==0 && N%BN==0 (enables async LDS staging).
//   HB/HR/O32/O16: compile-time epilogue configuration.
template<bool WF16, int ACT, bool GUARD, bool HB, bool HR, bool O32, bool O16>
__global__ void __launch_bounds__(256)
gemm_wmma(const half_t* __restrict__ A, int lda, long long sA,
          const void* __restrict__ Wp, int ldw, long long sW,
          const float* __restrict__ bias,
          const float* __restrict__ res, long long sR,
          float* __restrict__ C32, half_t* __restrict__ C16,
          int ldc, long long sC,
          int M, int N, int K, float scale)
{
  __shared__ __align__(16) half_t As[BM * LDT];
  __shared__ __align__(16) half_t Bs[BN * LDT];

  const int z = blockIdx.z;
  A += (size_t)z * sA;
  const float*  W32 = (const float*)Wp + (WF16 ? 0 : (size_t)z * sW);
  const half_t* W16 = (const half_t*)Wp + (WF16 ? (size_t)z * sW : 0);

  const int tid   = threadIdx.x;
  const int wave  = tid >> 5;
  const int lane  = tid & 31;
  const int wm    = wave >> 2;            // 0..1 -> 64 rows each
  const int wn    = wave & 3;             // 0..3 -> 32 cols each
  const int lhalf = lane & 15;
  const int khi   = (lane & 16) ? 8 : 0;
  const int row0  = blockIdx.y * BM;
  const int col0  = blockIdx.x * BN;

  v8f acc[4][2];
  const v8f z8 = {0.f, 0.f, 0.f, 0.f, 0.f, 0.f, 0.f, 0.f};
#pragma unroll
  for (int mi = 0; mi < 4; ++mi)
#pragma unroll
    for (int ni = 0; ni < 2; ++ni) acc[mi][ni] = z8;

  for (int k0 = 0; k0 < K; k0 += BK) {
    __syncthreads();
    // ---- stage A tile (BM x BK) : 2 x 16B chunks per thread ----
#pragma unroll
    for (int c = 0; c < 2; ++c) {
      int chunk = tid * 2 + c;
      int r     = chunk >> 2;
      int off   = (chunk & 3) * 8;
      int gr    = row0 + r;
      half_t* dst = &As[r * LDT + off];
      const half_t* ap = A + (size_t)gr * lda + k0 + off;
      if constexpr (!GUARD) {
        async_copy_b128(dst, ap);                 // global -> LDS, no VGPR trip
      } else {
        v8h v;
#pragma unroll
        for (int i = 0; i < 8; ++i) v[i] = (half_t)0.f;
        if (gr < M) v = *(const v8h*)(ap);
        *(v8h*)(dst) = v;
      }
    }
    // ---- stage W tile (BN x BK) ----
#pragma unroll
    for (int c = 0; c < 2; ++c) {
      int chunk = tid * 2 + c;
      int r     = chunk >> 2;             // output column within tile
      int off   = (chunk & 3) * 8;
      int gn    = col0 + r;
      half_t* dst = &Bs[r * LDT + off];
      if constexpr (WF16 && !GUARD) {
        async_copy_b128(dst, W16 + (size_t)gn * ldw + k0 + off);
      } else if constexpr (WF16) {        // guarded f16 W
        v8h v;
#pragma unroll
        for (int i = 0; i < 8; ++i) v[i] = (half_t)0.f;
        if (gn < N) v = *(const v8h*)(W16 + (size_t)gn * ldw + k0 + off);
        *(v8h*)(dst) = v;
      } else {                            // fp32 W: convert to f16 in-flight
        v8h v;
#pragma unroll
        for (int i = 0; i < 8; ++i) v[i] = (half_t)0.f;
        if (!GUARD || gn < N) {
          const float* wp = W32 + (size_t)gn * ldw + k0 + off;
          v4f lo = *(const v4f*)(wp);
          v4f hi = *(const v4f*)(wp + 4);
#pragma unroll
          for (int i = 0; i < 4; ++i) { v[i] = (half_t)lo[i]; v[4 + i] = (half_t)hi[i]; }
          if (k0 + BK < K) __builtin_prefetch(wp + BK, 0, 3);
        }
        *(v8h*)(dst) = v;
      }
    }
    if constexpr (!GUARD) wait_asynccnt0();       // our async writes landed in LDS
    __syncthreads();                              // ... and everyone else's

    v16h af[4], bf[2];
#pragma unroll
    for (int mi = 0; mi < 4; ++mi)
      af[mi] = ld_frag(&As[(wm * 64 + mi * 16 + lhalf) * LDT + khi]);
#pragma unroll
    for (int ni = 0; ni < 2; ++ni)
      bf[ni] = ld_frag(&Bs[(wn * 32 + ni * 16 + lhalf) * LDT + khi]);
#pragma unroll
    for (int mi = 0; mi < 4; ++mi)
#pragma unroll
      for (int ni = 0; ni < 2; ++ni)
        acc[mi][ni] = __builtin_amdgcn_wmma_f32_16x16x32_f16(
            false, af[mi], false, bf[ni], (short)0, acc[mi][ni], false, false);
  }

  // ---- epilogue: D layout = 8 VGPRs, M = r (+8 upper lane group), N = lane%16
  const int rlane = (lane & 16) ? 8 : 0;
  const int nlane = lane & 15;
#pragma unroll
  for (int ni = 0; ni < 2; ++ni) {
    const int col = col0 + wn * 32 + ni * 16 + nlane;
    if (GUARD && col >= N) continue;
    const float bv = HB ? bias[col] : 0.f;
#pragma unroll
    for (int mi = 0; mi < 4; ++mi) {
      const int rbase = row0 + wm * 64 + mi * 16 + rlane;
      const size_t ci = (size_t)z * sC + (size_t)rbase * ldc + col;
      const size_t ri = (size_t)z * sR + (size_t)rbase * ldc + col;
#pragma unroll
      for (int r = 0; r < 8; ++r) {
        if (GUARD && rbase + r >= M) continue;
        float v = acc[mi][ni][r] * scale + bv;
        if (ACT == 1) v = v / (1.f + __expf(-1.702f * v));   // quickGELU
        if (HR) v += res[ri + (size_t)r * ldc];
        if (O32) C32[ci + (size_t)r * ldc] = v;
        if (O16) C16[ci + (size_t)r * ldc] = (half_t)v;
      }
    }
  }
}

// ---------------- LayerNorm over D=768, one block per (b, t) row ------------
__global__ void __launch_bounds__(256)
ln_kernel(const float* __restrict__ X, const float* __restrict__ w,
          const float* __restrict__ bsh,
          float* __restrict__ Y32, half_t* __restrict__ Y16)
{
  __shared__ float red[256];
  const int row = blockIdx.x;
  const int t   = row % NTP;
  const int tid = threadIdx.x;
  const size_t base = (size_t)row * D_;
  if (t >= NV) {                       // padded token rows -> zeros
    for (int i = tid; i < D_; i += 256) {
      if (Y32) Y32[base + i] = 0.f;
      if (Y16) Y16[base + i] = (half_t)0.f;
    }
    return;
  }
  float lx[3];
  float s = 0.f;
#pragma unroll
  for (int j = 0; j < 3; ++j) { lx[j] = X[base + tid + j * 256]; s += lx[j]; }
  red[tid] = s; __syncthreads();
  for (int st = 128; st > 0; st >>= 1) { if (tid < st) red[tid] += red[tid + st]; __syncthreads(); }
  const float mean = red[0] * (1.f / (float)D_);
  __syncthreads();
  float s2 = 0.f;
#pragma unroll
  for (int j = 0; j < 3; ++j) { float d = lx[j] - mean; s2 += d * d; }
  red[tid] = s2; __syncthreads();
  for (int st = 128; st > 0; st >>= 1) { if (tid < st) red[tid] += red[tid + st]; __syncthreads(); }
  const float rstd = rsqrtf(red[0] * (1.f / (float)D_) + 1e-5f);
#pragma unroll
  for (int j = 0; j < 3; ++j) {
    int i = tid + j * 256;
    float y = (lx[j] - mean) * rstd * w[i] + bsh[i];
    if (Y32) Y32[base + i] = y;
    if (Y16) Y16[base + i] = (half_t)y;
  }
}

// ------------- softmax over keys; NACLIP gaussian addition inline -----------
__global__ void __launch_bounds__(256)
softmax_kernel(const float* __restrict__ scores, half_t* __restrict__ probs, int naclip)
{
  __shared__ float red[256];
  const int row = blockIdx.x;          // (b*H + h)*NTP + q
  const int q   = row % NTP;
  const int tid = threadIdx.x;
  const size_t base = (size_t)row * NTP;
  if (q >= NV) { if (tid < NTP) probs[base + tid] = (half_t)0.f; return; }
  const int kk = tid;
  const bool act = (kk < NV);
  float v = -3.0e38f;
  if (act) {
    v = scores[base + kk];
    if (naclip && q >= 1 && kk >= 1) {
      int qi = q - 1, ki = kk - 1;
      float dr = (float)((ki / 14) - (qi / 14));
      float dc = (float)((ki % 14) - (qi % 14));
      v += __expf(-0.02f * (dr * dr + dc * dc));   // 1/(2*STD^2) = 0.02
    }
  }
  red[tid] = v; __syncthreads();
  for (int st = 128; st > 0; st >>= 1) { if (tid < st) red[tid] = fmaxf(red[tid], red[tid + st]); __syncthreads(); }
  const float mx = red[0]; __syncthreads();
  const float e = act ? __expf(v - mx) : 0.f;
  red[tid] = e; __syncthreads();
  for (int st = 128; st > 0; st >>= 1) { if (tid < st) red[tid] += red[tid + st]; __syncthreads(); }
  const float inv = 1.f / red[0];
  if (tid < NTP) probs[base + tid] = (half_t)(e * inv);
}

// ---------------------------- elementwise kernels ---------------------------
__global__ void im2col_kernel(const float* __restrict__ x, half_t* __restrict__ patches)
{
  size_t idx = (size_t)blockIdx.x * 256 + threadIdx.x;
  if (idx >= (size_t)BT * D_) return;
  int kidx = (int)(idx % D_);
  int row  = (int)(idx / D_);
  int t = row % NTP, b = row / NTP;
  float v = 0.f;
  if (t >= 1 && t < NV) {
    int pidx = t - 1;
    int py = pidx / 14, px = pidx % 14;
    int c = kidx >> 8, rem = kidx & 255;
    int i = rem >> 4, j = rem & 15;
    v = x[(((size_t)b * 3 + c) * 224 + (py * 16 + i)) * 224 + (px * 16 + j)];
  }
  patches[idx] = (half_t)v;
}

__global__ void addemb_kernel(float* __restrict__ h, const float* __restrict__ cls,
                              const float* __restrict__ pos)
{
  size_t idx = (size_t)blockIdx.x * 256 + threadIdx.x;
  if (idx >= (size_t)BT * D_) return;
  int d = (int)(idx % D_);
  int row = (int)(idx / D_);
  int t = row % NTP;
  if (t >= NV) { h[idx] = 0.f; return; }
  float v = (t == 0) ? cls[d] : h[idx];
  h[idx] = v + pos[(size_t)t * D_ + d];
}

// qkv [B,NTP,2304] fp32 -> qh/kh [B,H,NTP,64] f16, vT [B,H,64,NTP] f16
__global__ void split_heads_kernel(const float* __restrict__ qkv,
                                   half_t* __restrict__ qh, half_t* __restrict__ kh,
                                   half_t* __restrict__ vT)
{
  size_t idx = (size_t)blockIdx.x * 256 + threadIdx.x;
  if (idx >= (size_t)BT * D_) return;
  int d  = (int)(idx % HD);
  size_t tmp = idx / HD;
  int hh = (int)(tmp % H_); tmp /= H_;
  int t  = (int)(tmp % NTP);
  int b  = (int)(tmp / NTP);
  size_t qi = ((size_t)(b * NTP + t)) * (3 * D_) + hh * HD + d;
  size_t ho = (((size_t)(b * H_ + hh)) * NTP + t) * HD + d;
  qh[ho] = (half_t)qkv[qi];
  kh[ho] = (half_t)qkv[qi + D_];
  vT[(((size_t)(b * H_ + hh)) * HD + d) * NTP + t] = (half_t)qkv[qi + 2 * D_];
}

// attnhead [B,H,NTP,64] f16 -> merged [B,NTP,768] f16
__global__ void merge_heads_kernel(const half_t* __restrict__ ah, half_t* __restrict__ am)
{
  size_t idx = (size_t)blockIdx.x * 256 + threadIdx.x;
  if (idx >= (size_t)BT * D_) return;
  int d  = (int)(idx % D_);
  int row = (int)(idx / D_);
  int t = row % NTP, b = row / NTP;
  int hh = d / HD, dd = d % HD;
  am[idx] = ah[(((size_t)(b * H_ + hh)) * NTP + t) * HD + dd];
}

// proj (768,512) fp32 -> projT (512,768) f16
__global__ void transpose_proj_kernel(const float* __restrict__ proj, half_t* __restrict__ projT)
{
  size_t idx = (size_t)blockIdx.x * 256 + threadIdx.x;
  if (idx >= (size_t)OUT_ * D_) return;
  int k = (int)(idx % D_);
  int n = (int)(idx / D_);
  projT[idx] = (half_t)proj[(size_t)k * OUT_ + n];
}

// finalbuf [B,NTP,512] -> d_out [B,197,512]
__global__ void pack_out_kernel(const float* __restrict__ fin, float* __restrict__ out)
{
  size_t idx = (size_t)blockIdx.x * 256 + threadIdx.x;
  if (idx >= (size_t)B_ * NV * OUT_) return;
  int o = (int)(idx % OUT_);
  size_t tmp = idx / OUT_;
  int t = (int)(tmp % NV);
  int b = (int)(tmp / NV);
  out[idx] = fin[((size_t)(b * NTP + t)) * OUT_ + o];
}

// ---------------------------------------------------------------------------
static inline dim3 g1(size_t n) { return dim3((unsigned)((n + 255) / 256)); }

// Epilogue/layout configurations actually used (compile-time dispatch):
//   cfg 0: fp32-W, full, C32                (patch embed)
//   cfg 1: fp32-W, full, bias, C32          (qkv)
//   cfg 2: fp32-W, full, bias, res, C32     (out_proj, c_proj)
//   cfg 3: fp32-W, full, bias, gelu, C16    (mlp fc)
//   cfg 4: f16-W, guarded, C32              (scores)
//   cfg 5: f16-W, guarded, C16              (probs @ V)
//   cfg 6: f16-W, full, C32                 (final projection)
static void gemm_cfg(int cfg, hipStream_t stream,
                     const half_t* A, int lda, long long sA,
                     const void* W, int ldw, long long sW,
                     const float* bias,
                     const float* res, long long sR,
                     float* C32, half_t* C16, int ldc, long long sC,
                     int M, int N, int K, float scale, int batch)
{
  dim3 grid((N + BN - 1) / BN, (M + BM - 1) / BM, batch);
#define GEMM_LAUNCH(WF, AC, GU, HB, HR, O32, O16)                                      \
  gemm_wmma<WF, AC, GU, HB, HR, O32, O16><<<grid, 256, 0, stream>>>(                   \
      A, lda, sA, W, ldw, sW, bias, res, sR, C32, C16, ldc, sC, M, N, K, scale)
  switch (cfg) {
    case 0: GEMM_LAUNCH(false, 0, false, false, false, true,  false); break;
    case 1: GEMM_LAUNCH(false, 0, false, true,  false, true,  false); break;
    case 2: GEMM_LAUNCH(false, 0, false, true,  true,  true,  false); break;
    case 3: GEMM_LAUNCH(false, 1, false, true,  false, false, true ); break;
    case 4: GEMM_LAUNCH(true,  0, true,  false, false, true,  false); break;
    case 5: GEMM_LAUNCH(true,  0, true,  false, false, false, true ); break;
    case 6: GEMM_LAUNCH(true,  0, false, false, false, true,  false); break;
    default: break;
  }
#undef GEMM_LAUNCH
}

extern "C" void kernel_launch(void* const* d_in, const int* in_sizes, int n_in,
                              void* d_out, int out_size, void* d_ws, size_t ws_size,
                              hipStream_t stream)
{
  (void)in_sizes; (void)n_in; (void)out_size; (void)ws_size;
  const float* x      = (const float*)d_in[0];
  const float* conv1w = (const float*)d_in[1];   // (768, 3*16*16) row-major == W[n][k]
  const float* clse   = (const float*)d_in[2];
  const float* pos    = (const float*)d_in[3];
  const float* lnprew = (const float*)d_in[4];
  const float* lnpreb = (const float*)d_in[5];
  const float* inpw   = (const float*)d_in[6];
  const float* inpb   = (const float*)d_in[7];
  const float* outpw  = (const float*)d_in[8];
  const float* outpb  = (const float*)d_in[9];
  const float* ln1w   = (const float*)d_in[10];
  const float* ln1b   = (const float*)d_in[11];
  const float* ln2w   = (const float*)d_in[12];
  const float* ln2b   = (const float*)d_in[13];
  const float* fcw    = (const float*)d_in[14];
  const float* fcb    = (const float*)d_in[15];
  const float* cpw    = (const float*)d_in[16];
  const float* cpb    = (const float*)d_in[17];
  const float* lnpow  = (const float*)d_in[18];
  const float* lnpob  = (const float*)d_in[19];
  const float* proj   = (const float*)d_in[20];
  float* out = (float*)d_out;

  // ---- workspace carving (~340 MB total) ----
  char* p = (char*)d_ws;
  auto carve = [&](size_t bytes) -> void* {
    void* r = (void*)p; p += (bytes + 255) & ~(size_t)255; return r;
  };
  float*  h        = (float*) carve((size_t)BT * D_ * 4);
  half_t* xln      = (half_t*)carve((size_t)BT * D_ * 2);
  half_t* patches  = (half_t*)carve((size_t)BT * D_ * 2);
  float*  qkv      = (float*) carve((size_t)BT * 3 * D_ * 4);
  half_t* qh       = (half_t*)carve((size_t)B_ * H_ * NTP * HD * 2);
  half_t* kh       = (half_t*)carve((size_t)B_ * H_ * NTP * HD * 2);
  half_t* vT       = (half_t*)carve((size_t)B_ * H_ * HD * NTP * 2);
  float*  scoresb  = (float*) carve((size_t)B_ * H_ * NTP * NTP * 4);
  half_t* probs    = (half_t*)carve((size_t)B_ * H_ * NTP * NTP * 2);
  half_t* ahead    = (half_t*)carve((size_t)B_ * H_ * NTP * HD * 2);
  half_t* amerge   = (half_t*)carve((size_t)BT * D_ * 2);
  half_t* mlpb     = (half_t*)carve((size_t)BT * F_ * 2);
  half_t* projT    = (half_t*)carve((size_t)OUT_ * D_ * 2);
  float*  finalbuf = (float*) carve((size_t)BT * OUT_ * 4);

  const long long ZH   = (long long)B_ * H_;             // 384 attention batches
  const long long sQK  = (long long)NTP * HD;            // qh/kh per-(b,h) stride
  const long long sVT  = (long long)HD * NTP;            // vT per-(b,h) stride
  const long long sSC  = (long long)NTP * NTP;           // scores/probs stride

  // ---- patch embedding: im2col + GEMM vs conv weight, then cls/pos add ----
  im2col_kernel<<<g1((size_t)BT * D_), 256, 0, stream>>>(x, patches);
  gemm_cfg(0, stream, patches, D_, 0, conv1w, D_, 0, nullptr, nullptr, 0,
           h, nullptr, D_, 0, BT, D_, D_, 1.f, 1);
  addemb_kernel<<<g1((size_t)BT * D_), 256, 0, stream>>>(h, clse, pos);

  // ---- ln_pre (in-place on the fp32 stream) ----
  ln_kernel<<<BT, 256, 0, stream>>>(h, lnprew, lnpreb, h, nullptr);

  // ---- transformer layers ----
  for (int i = 0; i < L_; ++i) {
    const bool naclip = (i == L_ - 1);

    // ln1 -> f16
    ln_kernel<<<BT, 256, 0, stream>>>(h, ln1w + (size_t)i * D_, ln1b + (size_t)i * D_,
                                      nullptr, xln);
    // qkv = xln @ in_proj_w[i]^T + b
    gemm_cfg(1, stream, xln, D_, 0,
             inpw + (size_t)i * 3 * D_ * D_, D_, 0, inpb + (size_t)i * 3 * D_,
             nullptr, 0, qkv, nullptr, 3 * D_, 0, BT, 3 * D_, D_, 1.f, 1);
    split_heads_kernel<<<g1((size_t)BT * D_), 256, 0, stream>>>(qkv, qh, kh, vT);

    // scores = (q|k) @ k^T * hd^-0.5   (batched over B*H)
    gemm_cfg(4, stream, naclip ? kh : qh, HD, sQK,
             kh, HD, sQK, nullptr, nullptr, 0,
             scoresb, nullptr, NTP, sSC, NTP, NTP, HD, 0.125f, (int)ZH);
    softmax_kernel<<<(unsigned)(ZH * NTP), 256, 0, stream>>>(scoresb, probs, naclip ? 1 : 0);

    // attn = probs @ v  (V staged transposed, batched)
    gemm_cfg(5, stream, probs, NTP, sSC,
             vT, NTP, sVT, nullptr, nullptr, 0,
             nullptr, ahead, HD, sQK, NTP, HD, NTP, 1.f, (int)ZH);
    merge_heads_kernel<<<g1((size_t)BT * D_), 256, 0, stream>>>(ahead, amerge);

    // h = h + attn @ out_proj_w[i]^T + b
    gemm_cfg(2, stream, amerge, D_, 0,
             outpw + (size_t)i * D_ * D_, D_, 0, outpb + (size_t)i * D_,
             h, 0, h, nullptr, D_, 0, BT, D_, D_, 1.f, 1);

    // ln2 -> f16 ; mlp hidden = quickgelu(xln @ fc^T + b) -> f16
    ln_kernel<<<BT, 256, 0, stream>>>(h, ln2w + (size_t)i * D_, ln2b + (size_t)i * D_,
                                      nullptr, xln);
    gemm_cfg(3, stream, xln, D_, 0,
             fcw + (size_t)i * F_ * D_, D_, 0, fcb + (size_t)i * F_,
             nullptr, 0, nullptr, mlpb, F_, 0, BT, F_, D_, 1.f, 1);
    // h = h + hidden @ cproj^T + b
    gemm_cfg(2, stream, mlpb, F_, 0,
             cpw + (size_t)i * D_ * F_, F_, 0, cpb + (size_t)i * D_,
             h, 0, h, nullptr, D_, 0, BT, D_, F_, 1.f, 1);
  }

  // ---- ln_post, final projection, pack valid tokens ----
  ln_kernel<<<BT, 256, 0, stream>>>(h, lnpow, lnpob, nullptr, xln);
  transpose_proj_kernel<<<g1((size_t)OUT_ * D_), 256, 0, stream>>>(proj, projT);
  gemm_cfg(6, stream, xln, D_, 0,
           projT, D_, 0, nullptr, nullptr, 0,
           finalbuf, nullptr, OUT_, 0, BT, OUT_, D_, 1.f, 1);
  pack_out_kernel<<<g1((size_t)B_ * NV * OUT_), 256, 0, stream>>>(finalbuf, out);
}
